// MemoryBank_8993661518606
// MI455X (gfx1250) — matrix operand order repaired and checked by reading, not compiled, not run on hardware
//
#include <hip/hip_runtime.h>
#include <stdint.h>

typedef __attribute__((ext_vector_type(16))) __bf16 v16bf;
typedef __attribute__((ext_vector_type(8)))  __bf16 v8bf;
typedef __attribute__((ext_vector_type(8)))  float  v8f;
typedef __attribute__((ext_vector_type(4)))  float  v4f;

#define B_    2
#define T_    2048
#define M_    8192
#define D_    1024
#define H_    8
#define DH_   128
#define TOPK_ 64

union frag16 { v16bf v; v8bf h[2]; };

// ---------- helpers ----------
__device__ __forceinline__ __bf16 f2bf(float f) {
  uint32_t u = __builtin_bit_cast(uint32_t, f);
  u += 0x7FFFu + ((u >> 16) & 1u);          // round-to-nearest-even
  uint16_t h = (uint16_t)(u >> 16);
  return __builtin_bit_cast(__bf16, h);
}

__device__ __forceinline__ float rowmax16(float v) {
#pragma unroll
  for (int off = 1; off < 16; off <<= 1) v = fmaxf(v, __shfl_xor(v, off, 32));
  return v;
}
__device__ __forceinline__ float rowsum16(float v) {
#pragma unroll
  for (int off = 1; off < 16; off <<= 1) v += __shfl_xor(v, off, 32);
  return v;
}

// ---------- kernel 0a: elementwise f32 -> bf16 (bandwidth-bound, one-shot) ----------
__global__ void __launch_bounds__(256)
cvt_bf16_kernel(const float* __restrict__ in, __bf16* __restrict__ out) {
  size_t i = ((size_t)blockIdx.x * 256 + threadIdx.x) * 4;
  v4f x = *(const v4f*)(in + i);
  out[i + 0] = f2bf(x[0]);
  out[i + 1] = f2bf(x[1]);
  out[i + 2] = f2bf(x[2]);
  out[i + 3] = f2bf(x[3]);
}

// ---------- kernel 0b: transpose + f32->bf16 weights (wt[c][k] = w[k][c]) ----------
__global__ void wtrans_kernel(const float* __restrict__ w, __bf16* __restrict__ wt) {
  int idx = blockIdx.x * 256 + threadIdx.x;
  int c = idx / D_;
  int k = idx % D_;
  wt[idx] = f2bf(w[(size_t)k * D_ + c]);
}

// ---------- kernel 1: projection  out[b,h,r,dh] = bf16((X@W + bias)*oscale) ----------
// X bf16 row-major [rows, D]; WT bf16 [D(out col), D(k)] row-major. RPB compile-time.
template <int RPB>
__global__ void __launch_bounds__(256)
proj_kernel(const __bf16* __restrict__ X, const __bf16* __restrict__ WT,
            const float* __restrict__ bias, __bf16* __restrict__ out, float oscale) {
  const int lane = threadIdx.x & 31;
  const int wave = threadIdx.x >> 5;
  const int tile = blockIdx.x * 8 + wave;
  const int trow = tile >> 6;            // D_/16 == 64 col tiles
  const int tcol = tile & 63;
  const int l15  = lane & 15;
  const int sel  = lane >> 4;

  const __bf16* xrow = X  + (size_t)(trow * 16 + l15) * D_;
  const __bf16* wrow = WT + (size_t)(tcol * 16 + l15) * D_;

  v8f c = {};
#pragma unroll 4
  for (int kb = 0; kb < D_; kb += 32) {
    frag16 a, b;
    a.h[0] = *(const v8bf*)(xrow + kb + sel * 8);
    a.h[1] = *(const v8bf*)(xrow + kb + 16 + sel * 8);
    b.h[0] = *(const v8bf*)(wrow + kb + sel * 8);
    b.h[1] = *(const v8bf*)(wrow + kb + 16 + sel * 8);
    c = __builtin_amdgcn_wmma_f32_16x16x32_bf16(false, a.v, false, b.v, (short)0,
                                                c, false, false);
  }

  const int col = tcol * 16 + l15;
  const int h   = col / DH_;
  const int kd  = col % DH_;
  const float bv = bias[col];
#pragma unroll
  for (int i = 0; i < 8; ++i) {
    int m  = trow * 16 + i + 8 * sel;        // global row
    int bb = m / RPB;                        // compile-time -> shifts
    int rl = m % RPB;
    size_t o = (((size_t)(bb * H_ + h) * RPB) + rl) * DH_ + kd;
    out[o] = f2bf((c[i] + bv) * oscale);
  }
}

// ---------- QK^T tile from LDS-staged K tile (16 rows x dh=128 bf16) ----------
__device__ __forceinline__ v8f qk_tile_lds(const v16bf* a, const __bf16* kt,
                                           int sel, int l15) {
  const __bf16* krow = kt + l15 * DH_ + sel * 8;
  v8f c = {};
#pragma unroll
  for (int s = 0; s < 4; ++s) {
    frag16 bf;
    bf.h[0] = *(const v8bf*)(krow + s * 32);      // ds_load_b128
    bf.h[1] = *(const v8bf*)(krow + s * 32 + 16);
    c = __builtin_amdgcn_wmma_f32_16x16x32_bf16(false, a[s], false, bf.v, (short)0,
                                                c, false, false);
  }
  return c;
}

// ---------- kernel 2: fused attention softmax + column accumulation ----------
// grid = B*H*(T/128) = 256 blocks; block owns one (b,h) and 128 query rows
// (wave w -> t-tile w). All 8 waves share each 4KB K tile via double-buffered
// LDS staging: K per batch streams from L2 once per block (8x less traffic).
__global__ void __launch_bounds__(256)
attn_kernel(const __bf16* __restrict__ q, const __bf16* __restrict__ k,
            float* __restrict__ acc) {
  __shared__ float  lacc[M_];                 // 32 KB column accumulator
  __shared__ __bf16 kt[2][16 * DH_];          // 2 x 4 KB K tile double buffer
  const int tid  = threadIdx.x;
  const int lane = tid & 31;
  const int wave = tid >> 5;
  const int bh   = blockIdx.x >> 4;           // T_/128 == 16 t-groups
  const int tg   = blockIdx.x & 15;
  const int b    = bh >> 3;                   // H_ == 8
  const int sel  = lane >> 4;
  const int l15  = lane & 15;

  for (int j = tid; j < M_; j += 256) lacc[j] = 0.0f;

  // A fragments: this wave's 16 q rows (t = tg*128 + wave*16 + ...), pinned
  const __bf16* qrow = q + (((size_t)bh * T_) + tg * 128 + wave * 16 + l15) * DH_;
  v16bf a[4];
#pragma unroll
  for (int s = 0; s < 4; ++s) {
    frag16 t;
    t.h[0] = *(const v8bf*)(qrow + s * 32 + sel * 8);
    t.h[1] = *(const v8bf*)(qrow + s * 32 + 16 + sel * 8);
    a[s] = t.v;
  }

  // staging addresses: thread -> (row srow of 16, 16B chunk schunk of 16)
  const __bf16* kbaseh = k + (size_t)bh * M_ * DH_;
  const int srow   = tid >> 4;
  const int schunk = tid & 15;
  const __bf16* gsrc = kbaseh + (size_t)srow * DH_ + schunk * 8;
  const int     loff = srow * DH_ + schunk * 8;

#define STAGE_LOAD(mt)  (*(const v8bf*)(gsrc + (size_t)(mt) * (16 * DH_)))
#define STAGE_STORE(bu, v) (*(v8bf*)(&kt[bu][loff]) = (v))

  float rmax[8], rsum[8], rinv[8];
#pragma unroll
  for (int i = 0; i < 8; ++i) { rmax[i] = -3.0e38f; rsum[i] = 0.0f; }

  // ---- phase 1: row max ----
  { v8bf v0 = STAGE_LOAD(0); STAGE_STORE(0, v0); }
  __syncthreads();
  for (int mt = 0; mt < M_ / 16; ++mt) {
    v8bf pre;
    const bool more = (mt + 1) < (M_ / 16);
    if (more) pre = STAGE_LOAD(mt + 1);
    v8f c = qk_tile_lds(a, &kt[mt & 1][0], sel, l15);
#pragma unroll
    for (int i = 0; i < 8; ++i) rmax[i] = fmaxf(rmax[i], c[i]);
    if (more) STAGE_STORE((mt + 1) & 1, pre);
    __syncthreads();
  }
#pragma unroll
  for (int i = 0; i < 8; ++i) rmax[i] = rowmax16(rmax[i]);

  // ---- phase 2: row sum of exp ----
  { v8bf v0 = STAGE_LOAD(0); STAGE_STORE(0, v0); }
  __syncthreads();
  for (int mt = 0; mt < M_ / 16; ++mt) {
    v8bf pre;
    const bool more = (mt + 1) < (M_ / 16);
    if (more) pre = STAGE_LOAD(mt + 1);
    v8f c = qk_tile_lds(a, &kt[mt & 1][0], sel, l15);
#pragma unroll
    for (int i = 0; i < 8; ++i) rsum[i] += __expf(c[i] - rmax[i]);
    if (more) STAGE_STORE((mt + 1) & 1, pre);
    __syncthreads();
  }
#pragma unroll
  for (int i = 0; i < 8; ++i) rinv[i] = 1.0f / rowsum16(rsum[i]);

  // ---- phase 3: normalize + column-sum into LDS (ds_add_f32 merges waves) ----
  { v8bf v0 = STAGE_LOAD(0); STAGE_STORE(0, v0); }
  __syncthreads();
  for (int mt = 0; mt < M_ / 16; ++mt) {
    v8bf pre;
    const bool more = (mt + 1) < (M_ / 16);
    if (more) pre = STAGE_LOAD(mt + 1);
    v8f c = qk_tile_lds(a, &kt[mt & 1][0], sel, l15);
    float cs = 0.0f;
#pragma unroll
    for (int i = 0; i < 8; ++i) cs += __expf(c[i] - rmax[i]) * rinv[i];
    cs += __shfl_xor(cs, 16, 32);            // fold the two row-halves
    if (lane < 16) atomicAdd(&lacc[mt * 16 + lane], cs);
    if (more) STAGE_STORE((mt + 1) & 1, pre);
    __syncthreads();
  }
#undef STAGE_LOAD
#undef STAGE_STORE

  for (int j = tid; j < M_; j += 256)
    atomicAdd(&acc[(size_t)b * M_ + j], lacc[j]);
}

// ---------- kernel 3: weighted = acc*scores, iterative top-64 + gather ----------
__global__ void __launch_bounds__(256)
topk_kernel(const float* __restrict__ acc, const float* __restrict__ scores,
            const float* __restrict__ mem, float* __restrict__ out) {
  __shared__ float w[M_];
  __shared__ float rv[256];
  __shared__ int   ri[256];
  __shared__ int   s_idx;
  const int b   = blockIdx.x;
  const int tid = threadIdx.x;

  for (int j = tid; j < M_; j += 256)
    w[j] = acc[(size_t)b * M_ + j] * scores[(size_t)b * M_ + j];
  __syncthreads();

  for (int it = 0; it < TOPK_; ++it) {
    float bv = -3.0e38f; int bi = 0x7fffffff;
    for (int j = tid; j < M_; j += 256)
      if (w[j] > bv || (w[j] == bv && j < bi)) { bv = w[j]; bi = j; }
    rv[tid] = bv; ri[tid] = bi;
    __syncthreads();
    for (int s = 128; s > 0; s >>= 1) {
      if (tid < s) {
        float ov = rv[tid + s]; int oi = ri[tid + s];
        if (ov > rv[tid] || (ov == rv[tid] && oi < ri[tid])) { rv[tid] = ov; ri[tid] = oi; }
      }
      __syncthreads();
    }
    if (tid == 0) { s_idx = ri[0]; w[ri[0]] = -3.0e38f; }
    __syncthreads();
    const int idx = s_idx;
    for (int c = tid; c < D_; c += 256)
      out[(((size_t)b * TOPK_) + it) * D_ + c] = mem[((size_t)b * M_ + idx) * D_ + c];
    __syncthreads();
  }
}

__global__ void zero_kernel(float* p, int n) {
  int i = blockIdx.x * 256 + threadIdx.x;
  if (i < n) p[i] = 0.0f;
}

// ---------- launch ----------
extern "C" void kernel_launch(void* const* d_in, const int* in_sizes, int n_in,
                              void* d_out, int out_size, void* d_ws, size_t ws_size,
                              hipStream_t stream) {
  const float* query  = (const float*)d_in[0];
  const float* memory = (const float*)d_in[1];
  const float* scores = (const float*)d_in[2];
  const float* wq     = (const float*)d_in[3];
  const float* bq     = (const float*)d_in[4];
  const float* wk     = (const float*)d_in[5];
  const float* bk     = (const float*)d_in[6];
  float* out = (float*)d_out;

  char* ws = (char*)d_ws;
  __bf16* wqT = (__bf16*)(ws);                         //  2 MB
  __bf16* wkT = (__bf16*)(ws + ((size_t)2  << 20));    //  2 MB
  __bf16* qb  = (__bf16*)(ws + ((size_t)4  << 20));    //  8 MB  [B,H,T,dh]
  __bf16* kb  = (__bf16*)(ws + ((size_t)12 << 20));    // 34 MB  [B,H,M,dh]
  __bf16* xq  = (__bf16*)(ws + ((size_t)46 << 20));    //  8 MB  bf16 query
  __bf16* xm  = (__bf16*)(ws + ((size_t)54 << 20));    // 34 MB  bf16 memory
  float*  acc = (float*) (ws + ((size_t)88 << 20));    // 64 KB  [B,M]

  cvt_bf16_kernel<<<(B_ * T_ * D_) / 1024, 256, 0, stream>>>(query,  xq);
  cvt_bf16_kernel<<<(B_ * M_ * D_) / 1024, 256, 0, stream>>>(memory, xm);
  wtrans_kernel<<<D_ * D_ / 256, 256, 0, stream>>>(wq, wqT);
  wtrans_kernel<<<D_ * D_ / 256, 256, 0, stream>>>(wk, wkT);

  const float qscale = 0.08838834764831845f;   // 1/sqrt(dh), folded into Q
  proj_kernel<T_><<<(B_ * T_ / 16) * (D_ / 16) / 8, 256, 0, stream>>>(xq, wqT, bq, qb, qscale);
  proj_kernel<M_><<<(B_ * M_ / 16) * (D_ / 16) / 8, 256, 0, stream>>>(xm, wkT, bk, kb, 1.0f);

  zero_kernel<<<(B_ * M_ + 255) / 256, 256, 0, stream>>>(acc, B_ * M_);
  attn_kernel<<<B_ * H_ * (T_ / 128), 256, 0, stream>>>(qb, kb, acc);
  topk_kernel<<<B_, 256, 0, stream>>>(acc, scores, memory, out);
}
